// RNN_38680475467996
// MI455X (gfx1250) — matrix operand order repaired and checked by reading, not compiled, not run on hardware
//
#include <hip/hip_runtime.h>
#include <cstdint>
#include <cstddef>

#define SEQ   128
#define BATCH 32
#define NVOC  32000
#define NIN   1024
#define NHID  1024
#define NG    4096   // 4*NHID

typedef __attribute__((ext_vector_type(16))) __bf16 v16bf;
typedef __attribute__((ext_vector_type(8)))  float  v8f;

union BF16x16 { uint4 q[2]; v16bf v; };

// ---------------- workspace layout (bytes) ----------------
constexpr size_t SZ_WP  = (size_t)NIN * NG * 2;          // packed W  (bf16)
constexpr size_t SZ_UP  = (size_t)NHID * NG * 2;         // packed U  (bf16)
constexpr size_t SZ_DW  = (size_t)NVOC * NHID * 2;       // dec_w     (bf16)
constexpr size_t SZ_EMB = (size_t)SEQ * BATCH * NIN * 2; // embeddings(bf16)
constexpr size_t SZ_HS  = (size_t)SEQ * BATCH * NHID * 2;// all h_t   (bf16)
constexpr size_t SZ_HF  = (size_t)BATCH * NHID * 4;      // h fp32
constexpr size_t SZ_CF  = (size_t)BATCH * NHID * 4;      // c fp32
constexpr size_t SZ_HB  = (size_t)BATCH * NHID * 2;      // h bf16 (x2 ping-pong)

constexpr size_t OFF_WP  = 0;
constexpr size_t OFF_UP  = OFF_WP  + SZ_WP;
constexpr size_t OFF_DW  = OFF_UP  + SZ_UP;
constexpr size_t OFF_EMB = OFF_DW  + SZ_DW;
constexpr size_t OFF_HS  = OFF_EMB + SZ_EMB;
constexpr size_t OFF_HF  = OFF_HS  + SZ_HS;
constexpr size_t OFF_CF  = OFF_HF  + SZ_HF;
constexpr size_t OFF_HB0 = OFF_CF  + SZ_CF;
constexpr size_t OFF_HB1 = OFF_HB0 + SZ_HB;

// ---------------- helpers ----------------
__device__ __forceinline__ uint16_t f2bf(float f) {
  union { float f; uint32_t u; } x; x.f = f;
  uint32_t u = x.u;
  return (uint16_t)((u + 0x7FFFu + ((u >> 16) & 1u)) >> 16);  // RNE
}

__device__ __forceinline__ float sigm(float x) {
  return 1.0f / (1.0f + __expf(-x));
}

// A operand (16x32 bf16) from row-major [M x K]:
// lane L: m = L%16, VGPR0-7 hold k = kb..kb+15, kb=(L/16)*16
__device__ __forceinline__ v16bf load_a_tile(const uint16_t* __restrict__ A,
                                             int row_stride, int m0, int k0, int lane) {
  int m  = m0 + (lane & 15);
  int kb = k0 + ((lane >> 4) << 4);
  const uint16_t* p = A + (size_t)m * row_stride + kb;
  BF16x16 t;
  t.q[0] = *(const uint4*)(p);        // k = kb .. kb+7
  t.q[1] = *(const uint4*)(p + 8);    // k = kb+8 .. kb+15
  return t.v;
}

// B operand (32x16 bf16) from pre-packed layout: ((nt*KT+kt)*32 + lane)*16 contiguous
__device__ __forceinline__ v16bf load_b_pack(const uint16_t* __restrict__ P,
                                             int nt, int kt, int KT, int lane) {
  const uint16_t* p = P + (((size_t)nt * KT + kt) * 32 + (size_t)lane) * 16;
  BF16x16 t;
  t.q[0] = *(const uint4*)(p);
  t.q[1] = *(const uint4*)(p + 8);
  return t.v;
}

// B operand from N-major [N x K] bf16 (dec_w layout): per-lane K is contiguous
__device__ __forceinline__ v16bf load_b_nmajor(const uint16_t* __restrict__ Bm,
                                               int row_stride, int n0, int k0, int lane) {
  int n  = n0 + (lane & 15);
  int kb = k0 + ((lane >> 4) << 4);
  const uint16_t* p = Bm + (size_t)n * row_stride + kb;
  BF16x16 t;
  t.q[0] = *(const uint4*)(p);
  t.q[1] = *(const uint4*)(p + 8);
  return t.v;
}

#define WMMA_BF16(a, b, c) \
  __builtin_amdgcn_wmma_f32_16x16x32_bf16(false, (a), false, (b), (short)0, (c), false, false)

// ---------------- prep kernels ----------------

// Repack [K x N] fp32 row-major into WMMA-B bf16 tiles.
__global__ void pack_b_kernel(const float* __restrict__ src, uint16_t* __restrict__ dst,
                              int K, int N) {
  int tid  = blockIdx.x * blockDim.x + threadIdx.x;
  int KT   = K >> 5;
  int lane = tid & 31;
  int tile = tid >> 5;                 // tile = nt*KT + kt
  if (tile >= (N >> 4) * KT) return;
  int kt = tile % KT, nt = tile / KT;
  int n  = (nt << 4) + (lane & 15);
  int kb = (kt << 5) + ((lane >> 4) << 4);
  uint16_t* d = dst + ((size_t)tile * 32 + (size_t)lane) * 16;
  #pragma unroll
  for (int e = 0; e < 16; ++e)
    d[e] = f2bf(src[(size_t)(kb + e) * N + n]);
}

__global__ void cvt_bf16_kernel(const float* __restrict__ src, uint16_t* __restrict__ dst,
                                int n) {
  int tid = blockIdx.x * blockDim.x + threadIdx.x;
  if (tid < n) dst[tid] = f2bf(src[tid]);
}

__global__ void embed_kernel(const int* __restrict__ ids, const float* __restrict__ ew,
                             uint16_t* __restrict__ emb, int total) {
  int tid = blockIdx.x * blockDim.x + threadIdx.x;
  if (tid >= total) return;
  int k  = tid & (NIN - 1);
  int sb = tid >> 10;
  emb[tid] = f2bf(ew[(size_t)ids[sb] * NIN + k]);
}

__global__ void init_states_kernel(const float* __restrict__ h0, const float* __restrict__ c0,
                                   float* __restrict__ hf, float* __restrict__ cf,
                                   uint16_t* __restrict__ hb0, int n) {
  int tid = blockIdx.x * blockDim.x + threadIdx.x;
  if (tid >= n) return;
  float h = h0[tid];
  hf[tid] = h; cf[tid] = c0[tid]; hb0[tid] = f2bf(h);
}

__global__ void copy_states_kernel(const float* __restrict__ hf, const float* __restrict__ cf,
                                   float* __restrict__ out, int n) {
  int tid = blockIdx.x * blockDim.x + threadIdx.x;
  if (tid >= n) return;
  out[tid] = hf[tid];
  out[n + tid] = cf[tid];
}

// ---------------- LSTM step: one wave = one 16-wide hidden column slice ----------------
// Computes all 4 gate tiles (i,f,o,g) for batch rows 0..31 -> lane-local cell update.
__global__ void __launch_bounds__(32) lstm_step_kernel(
    int t,
    const uint16_t* __restrict__ emb,   // [SEQ*BATCH, NIN]
    const uint16_t* __restrict__ h_in,  // [BATCH, NHID] bf16 (read)
    uint16_t*       __restrict__ h_out, // [BATCH, NHID] bf16 (write)
    const uint16_t* __restrict__ Wp,    // packed, KT = NIN/32
    const uint16_t* __restrict__ Up,    // packed, KT = NHID/32
    const float*    __restrict__ bias,  // [4*NHID]
    float*          __restrict__ h_f32,
    float*          __restrict__ c_f32,
    uint16_t*       __restrict__ hs_bf) // [SEQ*BATCH, NHID]
{
  const int lane = threadIdx.x;
  const int nt   = blockIdx.x;          // 0..63  (hidden cols nt*16 .. +15)
  const int KT   = NIN >> 5;            // 32
  const int ncol = (nt << 4) + (lane & 15);

  v8f acc[2][4];
  #pragma unroll
  for (int g = 0; g < 4; ++g) {
    float bv = bias[g * NHID + ncol];
    #pragma unroll
    for (int mt = 0; mt < 2; ++mt)
      #pragma unroll
      for (int r = 0; r < 8; ++r) acc[mt][g][r] = bv;
  }

  // gates += x_t @ W
  const uint16_t* x = emb + (size_t)t * BATCH * NIN;
  for (int kt = 0; kt < KT; ++kt) {
    v16bf a0 = load_a_tile(x, NIN, 0,  kt << 5, lane);
    v16bf a1 = load_a_tile(x, NIN, 16, kt << 5, lane);
    #pragma unroll
    for (int g = 0; g < 4; ++g) {
      v16bf b = load_b_pack(Wp, g * 64 + nt, kt, KT, lane);
      acc[0][g] = WMMA_BF16(a0, b, acc[0][g]);
      acc[1][g] = WMMA_BF16(a1, b, acc[1][g]);
    }
  }
  // gates += h @ U
  for (int kt = 0; kt < KT; ++kt) {
    v16bf a0 = load_a_tile(h_in, NHID, 0,  kt << 5, lane);
    v16bf a1 = load_a_tile(h_in, NHID, 16, kt << 5, lane);
    #pragma unroll
    for (int g = 0; g < 4; ++g) {
      v16bf b = load_b_pack(Up, g * 64 + nt, kt, KT, lane);
      acc[0][g] = WMMA_BF16(a0, b, acc[0][g]);
      acc[1][g] = WMMA_BF16(a1, b, acc[1][g]);
    }
  }

  // lane-local cell update:  element r of tile mt -> batch b = mt*16 + r + 8*(lane/16)
  #pragma unroll
  for (int mt = 0; mt < 2; ++mt) {
    #pragma unroll
    for (int r = 0; r < 8; ++r) {
      int b   = (mt << 4) + r + ((lane >> 4) << 3);
      int idx = b * NHID + ncol;
      float iv = sigm(acc[mt][0][r]);
      float fv = sigm(acc[mt][1][r]);
      float ov = sigm(acc[mt][2][r]);
      float gv = tanhf(acc[mt][3][r]);
      float cn = fv * c_f32[idx] + iv * gv;
      float hn = ov * tanhf(cn);
      c_f32[idx] = cn;
      h_f32[idx] = hn;
      uint16_t hb = f2bf(hn);
      h_out[idx] = hb;
      hs_bf[((size_t)t * BATCH + b) * NHID + ncol] = hb;
    }
  }
}

// ---------------- decoder GEMM: wave = 32x64 output tile, K=1024 ----------------
__global__ void __launch_bounds__(32) decoder_kernel(
    const uint16_t* __restrict__ hs,   // [SEQ*BATCH, NHID] bf16
    const uint16_t* __restrict__ dw,   // [NVOC, NHID] bf16 (N-major, K contiguous)
    const float*    __restrict__ db,   // [NVOC]
    float*          __restrict__ out)  // [SEQ*BATCH, NVOC]
{
  const int lane = threadIdx.x;
  const int nb   = blockIdx.x;         // 0..499 : 64 vocab cols
  const int mb   = blockIdx.y;         // 0..127 : 32 rows

  v8f acc[2][4];
  #pragma unroll
  for (int j = 0; j < 4; ++j) {
    float bv = db[nb * 64 + (j << 4) + (lane & 15)];
    #pragma unroll
    for (int mt = 0; mt < 2; ++mt)
      #pragma unroll
      for (int r = 0; r < 8; ++r) acc[mt][j][r] = bv;
  }

  for (int kt = 0; kt < (NHID >> 5); ++kt) {
    v16bf a0 = load_a_tile(hs, NHID, mb * 32,      kt << 5, lane);
    v16bf a1 = load_a_tile(hs, NHID, mb * 32 + 16, kt << 5, lane);
    #pragma unroll
    for (int j = 0; j < 4; ++j) {
      v16bf b = load_b_nmajor(dw, NHID, nb * 64 + (j << 4), kt << 5, lane);
      acc[0][j] = WMMA_BF16(a0, b, acc[0][j]);
      acc[1][j] = WMMA_BF16(a1, b, acc[1][j]);
    }
  }

  #pragma unroll
  for (int mt = 0; mt < 2; ++mt)
    #pragma unroll
    for (int j = 0; j < 4; ++j) {
      int n = nb * 64 + (j << 4) + (lane & 15);
      #pragma unroll
      for (int r = 0; r < 8; ++r) {
        int m = mb * 32 + (mt << 4) + r + ((lane >> 4) << 3);
        out[(size_t)m * NVOC + n] = acc[mt][j][r];
      }
    }
}

// ---------------- host launcher ----------------
extern "C" void kernel_launch(void* const* d_in, const int* in_sizes, int n_in,
                              void* d_out, int out_size, void* d_ws, size_t ws_size,
                              hipStream_t stream) {
  const int*   ids = (const int*)  d_in[0];
  const float* h0  = (const float*)d_in[1];
  const float* c0  = (const float*)d_in[2];
  const float* ew  = (const float*)d_in[3];
  const float* W   = (const float*)d_in[4];
  const float* U   = (const float*)d_in[5];
  const float* B   = (const float*)d_in[6];
  const float* dw  = (const float*)d_in[7];
  const float* db  = (const float*)d_in[8];
  float* out = (float*)d_out;

  char* ws = (char*)d_ws;
  uint16_t* Wp   = (uint16_t*)(ws + OFF_WP);
  uint16_t* Up   = (uint16_t*)(ws + OFF_UP);
  uint16_t* dwb  = (uint16_t*)(ws + OFF_DW);
  uint16_t* emb  = (uint16_t*)(ws + OFF_EMB);
  uint16_t* hsb  = (uint16_t*)(ws + OFF_HS);
  float*    hf   = (float*)   (ws + OFF_HF);
  float*    cf   = (float*)   (ws + OFF_CF);
  uint16_t* hb0  = (uint16_t*)(ws + OFF_HB0);
  uint16_t* hb1  = (uint16_t*)(ws + OFF_HB1);

  // 1) one-time prep (re-run every call for determinism)
  {
    int threads = (NG / 16) * (NIN / 32) * 32;   // 262144
    pack_b_kernel<<<threads / 256, 256, 0, stream>>>(W, Wp, NIN, NG);
    pack_b_kernel<<<threads / 256, 256, 0, stream>>>(U, Up, NHID, NG);
  }
  {
    int n = NVOC * NHID;                          // 32,768,000
    cvt_bf16_kernel<<<(n + 255) / 256, 256, 0, stream>>>(dw, dwb, n);
  }
  {
    int n = SEQ * BATCH * NIN;                    // 4,194,304
    embed_kernel<<<(n + 255) / 256, 256, 0, stream>>>(ids, ew, emb, n);
  }
  {
    int n = BATCH * NHID;                         // 32,768
    init_states_kernel<<<(n + 255) / 256, 256, 0, stream>>>(h0, c0, hf, cf, hb0, n);
  }

  // 2) sequential LSTM recurrence (64 waves/step, ping-pong h buffers)
  for (int t = 0; t < SEQ; ++t) {
    const uint16_t* hin  = (t & 1) ? hb1 : hb0;
    uint16_t*       hout = (t & 1) ? hb0 : hb1;
    lstm_step_kernel<<<NHID / 16, 32, 0, stream>>>(t, emb, hin, hout, Wp, Up, B,
                                                   hf, cf, hsb);
  }

  // 3) decoder GEMM (WMMA-compute bound)
  decoder_kernel<<<dim3(NVOC / 64, (SEQ * BATCH) / 32), 32, 0, stream>>>(hsb, dwb, db, out);

  // 4) append (h_t, c_t)
  {
    int n = BATCH * NHID;
    copy_states_kernel<<<(n + 255) / 256, 256, 0, stream>>>(
        hf, cf, out + (size_t)SEQ * BATCH * NVOC, n);
  }
}
